// MLA_90426241450568
// MI455X (gfx1250) — matrix operand order repaired and checked by reading, not compiled, not run on hardware
//
#include <hip/hip_runtime.h>

// Problem constants (match reference)
#define B_   2
#define S_   2048
#define H_   2048
#define NH_  16
#define DN_  128
#define DR_  64
#define DV_  128
#define QD_  192      // DN_ + DR_
#define QR_  1536
#define KVR_ 512

typedef __attribute__((ext_vector_type(16))) __bf16          v16bf;
typedef __attribute__((ext_vector_type(16))) unsigned short  v16us;
typedef __attribute__((ext_vector_type(8)))  unsigned short  v8us;
typedef __attribute__((ext_vector_type(8)))  float           v8f;
typedef __attribute__((ext_vector_type(4)))  unsigned int    v4u;
typedef __attribute__((ext_vector_type(8)))  unsigned int    v8u;

static __device__ __forceinline__ unsigned short f2bf_rn(float f) {
  unsigned int u = __float_as_uint(f);
  u += 0x7FFFu + ((u >> 16) & 1u);           // round-to-nearest-even
  return (unsigned short)(u >> 16);
}
static __device__ __forceinline__ float bf2f(unsigned short h) {
  return __uint_as_float(((unsigned int)h) << 16);
}
// Split 8 f32 into bf16 hi plane + bf16 residual (lo) plane.
static __device__ __forceinline__ void split8(const float* x, v8us& hi, v8us& lo) {
#pragma unroll
  for (int i = 0; i < 8; ++i) {
    unsigned short h = f2bf_rn(x[i]);
    hi[i] = h;
    lo[i] = f2bf_rn(x[i] - bf2f(h));
  }
}
// Build a 16-element bf16 fragment from two aligned 8-element LDS groups.
static __device__ __forceinline__ v16bf frag2(const unsigned short* p0,
                                              const unsigned short* p1) {
  v8us a = *(const v8us*)p0;
  v8us b = *(const v8us*)p1;
  v16us r;
#pragma unroll
  for (int i = 0; i < 8; ++i) { r[i] = a[i]; r[8 + i] = b[i]; }
  return __builtin_bit_cast(v16bf, r);
}

// ---------------------------------------------------------------------------
// TDM: issue a tensor DMA of a 128x32 f32 tile (row stride = lda elements)
// from global memory into LDS at lds_off, with LDS padding of 4 DWORDs every
// 32 DWORDs (effective pitch 36 floats). D# built per CDNA5 ISA §8.3/8.4.
// Group1 (geometry) is loop-invariant; group0 carries the per-tile addresses.
// ---------------------------------------------------------------------------
static __device__ __forceinline__ v8u tdm_group1(int K, int M, int lda) {
  v8u g1;
  // data_size=2 (4B) | pad_enable | pad_interval=4 (32 DW) | pad_amount=3 (4 DW)
  g1[0] = 0x07120000u;
  g1[1] = ((unsigned)K & 0xFFFFu) << 16;                               // tensor_dim0 lo
  g1[2] = (((unsigned)K >> 16) & 0xFFFFu) | (((unsigned)M & 0xFFFFu) << 16);
  g1[3] = (((unsigned)M >> 16) & 0xFFFFu) | (32u << 16);               // tile_dim0=32
  g1[4] = 128u;                                                        // tile_dim1=128
  g1[5] = (unsigned)lda;                                               // dim0 stride lo
  g1[6] = 0u;
  g1[7] = 0u;
  return g1;
}
static __device__ __forceinline__ void tdm_issue(const float* gtile,
                                                 unsigned lds_off, v8u g1) {
  unsigned long long ga = (unsigned long long)(const void*)gtile;
  v4u g0;
  g0[0] = 1u;                                          // count=1 (valid descriptor)
  g0[1] = lds_off;                                     // LDS byte address
  g0[2] = (unsigned)(ga & 0xFFFFFFFFu);                // global_addr[31:0]
  g0[3] = (unsigned)((ga >> 32) & 0x01FFFFFFu) | (2u << 30);  // addr[56:32], type=2
  asm volatile("tensor_load_to_lds %0, %1" :: "s"(g0), "s"(g1) : "memory");
}

// ---------------------------------------------------------------------------
// Generic batched WMMA GEMM: C[z] = alpha * A[z] (MxK) * B[z] (KxN)
// TRANSB=true  -> B is stored NxK row-major (computes A * B^T)
// z = blockIdx.z decomposed as (b, h) with per-matrix (b,h) element strides.
// A tiles arrive via double-buffered TDM (tensor_load_to_lds, TENSORcnt);
// bf16x3 split (hi*hi + hi*lo + lo*hi, f32 accumulate) done ONCE at staging
// into LDS bf16 hi/lo planes, so the hot loop is pure ds_load_b128 + v_wmma.
// Block tile 128(M) x 64(N), K-step 32. 8 waves (wave32): 4x2 grid of 32x32.
// ---------------------------------------------------------------------------
template <bool TRANSB>
__global__ __launch_bounds__(256)
void wmma_gemm(const float* __restrict__ Ag, const float* __restrict__ Bg,
               float* __restrict__ Cg,
               int M, int N, int K, int lda, int ldb, int ldc,
               long sAb, long sAh, long sBb, long sBh, long sCb, long sCh,
               int nh, float alpha)
{
  // f32 staging for TDM A tiles: pitch 36 floats (32 data + 4 pad DWORDs)
  __shared__ float sStage[2][128 * 36];
  // bf16 planes; pitch 40 ushorts (80B) keeps 16B alignment + spreads banks
  __shared__ unsigned short sAhi[128][40];
  __shared__ unsigned short sAlo[128][40];
  __shared__ unsigned short sBhi[64][40];    // [n][k] : K-contiguous per column
  __shared__ unsigned short sBlo[64][40];

  const int z  = blockIdx.z;
  const int bb = z / nh, hh = z % nh;
  const float* A  = Ag + (size_t)bb * sAb + (size_t)hh * sAh;
  const float* Bm = Bg + (size_t)bb * sBb + (size_t)hh * sBh;
  float*       C  = Cg + (size_t)bb * sCb + (size_t)hh * sCh;

  const int m0 = blockIdx.y * 128;
  const int n0 = blockIdx.x * 64;

  const int tid   = threadIdx.x;
  const int wid   = tid >> 5;
  const int lane  = tid & 31;
  const int half  = lane >> 4;      // K-half select per ISA A/B layouts
  const int l16   = lane & 15;
  const int waveM = (wid & 3) * 32;
  const int waveN = (wid >> 2) * 32;

  const v8f vzero = {0.f,0.f,0.f,0.f,0.f,0.f,0.f,0.f};
  v8f acc[2][2];
  acc[0][0] = vzero; acc[0][1] = vzero; acc[1][0] = vzero; acc[1][1] = vzero;

  // Per-thread staging coordinates (fixed across K loop)
  const int ar = tid >> 1;                 // A row 0..127
  const int ak = (tid & 1) * 16;           // A k-group 0 or 16
  const int bnT = tid >> 2;                // TRANSB: n 0..63
  const int bkT = (tid & 3) * 8;           // TRANSB: k 0,8,16,24
  const int bnN = tid & 63;                // normal: n 0..63
  const int bkN = (tid >> 6) * 8;          // normal: k 0,8,16,24

  const v8u g1 = tdm_group1(K, M, lda);
  const unsigned ldsStage0 = (unsigned)(unsigned long long)&sStage[0][0];
  const unsigned ldsStage1 = (unsigned)(unsigned long long)&sStage[1][0];
  const float* Atile0 = A + (size_t)m0 * lda;

  // prologue: wave 0 kicks off the DMA for the first A tile
  if (wid == 0) tdm_issue(Atile0, ldsStage0, g1);

  for (int k0 = 0; k0 < K; k0 += 32) {
    const int cur = (k0 >> 5) & 1;
    if (wid == 0) __builtin_amdgcn_s_wait_tensorcnt(0);
    __syncthreads();   // TDM(cur) visible; previous iteration's readers done

    // ---- convert A tile (f32 staging LDS -> bf16 hi/lo planes), once ----
    {
      const float* st = &sStage[cur][ar * 36 + ak];
      float x[16];
      *(float4*)&x[0]  = *(const float4*)(st + 0);
      *(float4*)&x[4]  = *(const float4*)(st + 4);
      *(float4*)&x[8]  = *(const float4*)(st + 8);
      *(float4*)&x[12] = *(const float4*)(st + 12);
      v8us h0, l0, h1, l1;
      split8(&x[0], h0, l0);
      split8(&x[8], h1, l1);
      *(v8us*)&sAhi[ar][ak]     = h0;  *(v8us*)&sAhi[ar][ak + 8] = h1;
      *(v8us*)&sAlo[ar][ak]     = l0;  *(v8us*)&sAlo[ar][ak + 8] = l1;
    }
    // ---- stage B tile (32x64) as bf16 hi/lo in [n][k] layout ----
    if (!TRANSB) {
      const float* gb = Bm + (size_t)(k0 + bkN) * ldb + (n0 + bnN);
      float x[8];
#pragma unroll
      for (int i = 0; i < 8; ++i) x[i] = gb[(size_t)i * ldb];
      v8us h, l;
      split8(x, h, l);
      *(v8us*)&sBhi[bnN][bkN] = h;
      *(v8us*)&sBlo[bnN][bkN] = l;
      if (k0 + 32 < K) __builtin_prefetch(gb + (size_t)32 * ldb, 0, 1);
    } else {
      const float* gb = Bm + (size_t)(n0 + bnT) * ldb + (k0 + bkT);
      float x[8];
      *(float4*)&x[0] = *(const float4*)(gb + 0);
      *(float4*)&x[4] = *(const float4*)(gb + 4);
      v8us h, l;
      split8(x, h, l);
      *(v8us*)&sBhi[bnT][bkT] = h;
      *(v8us*)&sBlo[bnT][bkT] = l;
      if (k0 + 32 < K) __builtin_prefetch(gb + 32, 0, 1);
    }
    // ---- kick off DMA for the next A tile into the other buffer ----
    if (wid == 0 && k0 + 32 < K)
      tdm_issue(Atile0 + (k0 + 32), cur ? ldsStage0 : ldsStage1, g1);
    __syncthreads();   // bf16 planes ready

    // ---- fragments: pure aligned ds_load_b128, no conversion ----
    v16bf aHi[2], aLo[2], bHi[2], bLo[2];
#pragma unroll
    for (int mt = 0; mt < 2; ++mt) {
      const int row = waveM + mt * 16 + l16;
      const int kb  = half * 8;            // lanes<16: K0..7/16..23, else 8..15/24..31
      aHi[mt] = frag2(&sAhi[row][kb], &sAhi[row][16 + kb]);
      aLo[mt] = frag2(&sAlo[row][kb], &sAlo[row][16 + kb]);
    }
#pragma unroll
    for (int nt = 0; nt < 2; ++nt) {
      const int col = waveN + nt * 16 + l16;   // N = lane%16
      const int kb  = half * 8;                // lanes<16: K0..7/16..23, else 8..15/24..31
      bHi[nt] = frag2(&sBhi[col][kb], &sBhi[col][16 + kb]);
      bLo[nt] = frag2(&sBlo[col][kb], &sBlo[col][16 + kb]);
    }

    // ---- bf16x3 WMMA ----
#pragma unroll
    for (int mt = 0; mt < 2; ++mt)
#pragma unroll
      for (int nt = 0; nt < 2; ++nt) {
        acc[mt][nt] = __builtin_amdgcn_wmma_f32_16x16x32_bf16(
            false, aHi[mt], false, bHi[nt], (short)0, acc[mt][nt], false, false);
        acc[mt][nt] = __builtin_amdgcn_wmma_f32_16x16x32_bf16(
            false, aHi[mt], false, bLo[nt], (short)0, acc[mt][nt], false, false);
        acc[mt][nt] = __builtin_amdgcn_wmma_f32_16x16x32_bf16(
            false, aLo[mt], false, bHi[nt], (short)0, acc[mt][nt], false, false);
      }
  }

  // ---- store C: VGPR r -> row r (lanes 0-15) / r+8 (lanes 16-31), N = lane%16 ----
#pragma unroll
  for (int mt = 0; mt < 2; ++mt)
#pragma unroll
    for (int nt = 0; nt < 2; ++nt) {
      const int baseM = m0 + waveM + mt * 16 + half * 8;
      const int baseN = n0 + waveN + nt * 16 + l16;
#pragma unroll
      for (int r = 0; r < 8; ++r)
        C[(size_t)(baseM + r) * ldc + baseN] = alpha * acc[mt][nt][r];
    }
}

// ---------------------------------------------------------------------------
// RMSNorm in-place over the first n columns of each `pitch`-wide row.
// ---------------------------------------------------------------------------
__global__ __launch_bounds__(256)
void rmsnorm_kernel(float* __restrict__ X, const float* __restrict__ w,
                    int pitch, int n)
{
  __shared__ float red[256];
  float* row = X + (size_t)blockIdx.x * pitch;
  const int t = threadIdx.x;
  float s = 0.f;
  for (int c = t; c < n; c += 256) { float v = row[c]; s += v * v; }
  red[t] = s; __syncthreads();
  for (int off = 128; off > 0; off >>= 1) {
    if (t < off) red[t] += red[t + off];
    __syncthreads();
  }
  const float scale = rsqrtf(red[0] / (float)n + 1e-6f);
  for (int c = t; c < n; c += 256) row[c] = row[c] * scale * w[c];
}

// ---------------------------------------------------------------------------
// Build per-head Q and K (nope copy + RoPE on last 64 dims). K-rope broadcast
// across heads. Qh/Kh layout: [B, NH, S, 192].
// ---------------------------------------------------------------------------
__global__ __launch_bounds__(128)
void rope_build_kernel(const float* __restrict__ T2, const float* __restrict__ T3,
                       const float* __restrict__ T4, const int* __restrict__ pos,
                       float* __restrict__ Qh, float* __restrict__ Kh)
{
  const int s = blockIdx.x, h = blockIdx.y, b = blockIdx.z, t = threadIdx.x;
  const size_t rowi = (size_t)b * S_ + s;
  const float* q  = T2 + rowi * (NH_ * QD_)         + (size_t)h * QD_;
  const float* kn = T4 + rowi * (NH_ * (DN_ + DV_)) + (size_t)h * (DN_ + DV_);
  const float* kr = T3 + rowi * (KVR_ + DR_)        + KVR_;
  const size_t o  = (((size_t)b * NH_ + h) * S_ + s) * QD_;
  float* Q  = Qh + o;
  float* Kd = Kh + o;

  Q[t]  = q[t];        // q_nope (t < 128)
  Kd[t] = kn[t];       // k_nope
  if (t < DR_ / 2) {
    const float p   = (float)pos[s];
    const float inv = powf(10000.f, -(float)(2 * t) / (float)DR_);
    const float a   = p * inv;
    const float c = cosf(a), sn = sinf(a);
    float x1 = q[DN_ + t], x2 = q[DN_ + 32 + t];
    Q[DN_ + t]       = x1 * c - x2 * sn;
    Q[DN_ + 32 + t]  = x2 * c + x1 * sn;
    float y1 = kr[t], y2 = kr[32 + t];
    Kd[DN_ + t]      = y1 * c - y2 * sn;
    Kd[DN_ + 32 + t] = y2 * c + y1 * sn;
  }
}

// ---------------------------------------------------------------------------
// Row softmax over width 2048, in place (numerically stable).
// ---------------------------------------------------------------------------
__global__ __launch_bounds__(256)
void softmax_kernel(float* __restrict__ attn)
{
  __shared__ float red[256];
  float* row = attn + (size_t)blockIdx.x * 2048;
  const int t = threadIdx.x;
  float v[8];
  float m = -3.402823466e38f;
#pragma unroll
  for (int i = 0; i < 8; ++i) { v[i] = row[t + i * 256]; m = fmaxf(m, v[i]); }
  red[t] = m; __syncthreads();
  for (int off = 128; off > 0; off >>= 1) {
    if (t < off) red[t] = fmaxf(red[t], red[t + off]);
    __syncthreads();
  }
  m = red[0]; __syncthreads();
  float s = 0.f;
#pragma unroll
  for (int i = 0; i < 8; ++i) { v[i] = __expf(v[i] - m); s += v[i]; }
  red[t] = s; __syncthreads();
  for (int off = 128; off > 0; off >>= 1) {
    if (t < off) red[t] += red[t + off];
    __syncthreads();
  }
  const float inv = 1.f / red[0];
#pragma unroll
  for (int i = 0; i < 8; ++i) row[t + i * 256] = v[i] * inv;
}

// ---------------------------------------------------------------------------
extern "C" void kernel_launch(void* const* d_in, const int* in_sizes, int n_in,
                              void* d_out, int out_size, void* d_ws, size_t ws_size,
                              hipStream_t stream)
{
  (void)in_sizes; (void)n_in; (void)out_size; (void)ws_size;
  const float* hidden    = (const float*)d_in[0];
  const int*   pos       = (const int*)d_in[1];
  const float* q_down_W  = (const float*)d_in[2];
  const float* q_norm_w  = (const float*)d_in[3];
  const float* q_up_W    = (const float*)d_in[4];
  const float* kv_down_W = (const float*)d_in[5];
  const float* kv_norm_w = (const float*)d_in[6];
  const float* kv_up_W   = (const float*)d_in[7];
  const float* out_W     = (const float*)d_in[8];

  float* out  = (float*)d_out;
  float* attn = out + (size_t)B_ * S_ * H_;     // [B, NH, S, S]

  float* p  = (float*)d_ws;
  float* T1 = p; p += (size_t)B_ * S_ * QR_;                 // q latent
  float* T2 = p; p += (size_t)B_ * S_ * NH_ * QD_;           // q up
  float* T3 = p; p += (size_t)B_ * S_ * (KVR_ + DR_);        // kv latent + k_rope
  float* T4 = p; p += (size_t)B_ * S_ * NH_ * (DN_ + DV_);   // kv up (k_nope|v)
  float* Qh = p; p += (size_t)B_ * NH_ * S_ * QD_;
  float* Kh = p; p += (size_t)B_ * NH_ * S_ * QD_;
  float* Cx = p; p += (size_t)B_ * S_ * NH_ * DV_;           // attention context

  const int R = B_ * S_;   // 4096 flattened rows
  dim3 blk(256);

  // 1) T1 = hidden @ q_down_W   [4096 x 1536]
  wmma_gemm<false><<<dim3(QR_ / 64, R / 128, 1), blk, 0, stream>>>(
      hidden, q_down_W, T1, R, QR_, H_, H_, QR_, QR_,
      0, 0, 0, 0, 0, 0, 1, 1.0f);
  // 2) rmsnorm(T1)
  rmsnorm_kernel<<<R, 256, 0, stream>>>(T1, q_norm_w, QR_, QR_);
  // 3) T2 = T1 @ q_up_W   [4096 x 3072]
  wmma_gemm<false><<<dim3((NH_ * QD_) / 64, R / 128, 1), blk, 0, stream>>>(
      T1, q_up_W, T2, R, NH_ * QD_, QR_, QR_, NH_ * QD_, NH_ * QD_,
      0, 0, 0, 0, 0, 0, 1, 1.0f);
  // 4) T3 = hidden @ kv_down_W   [4096 x 576]
  wmma_gemm<false><<<dim3((KVR_ + DR_) / 64, R / 128, 1), blk, 0, stream>>>(
      hidden, kv_down_W, T3, R, KVR_ + DR_, H_, H_, KVR_ + DR_, KVR_ + DR_,
      0, 0, 0, 0, 0, 0, 1, 1.0f);
  // 5) rmsnorm on first 512 cols of T3
  rmsnorm_kernel<<<R, 256, 0, stream>>>(T3, kv_norm_w, KVR_ + DR_, KVR_);
  // 6) T4 = T3[:, :512] @ kv_up_W   [4096 x 4096]
  wmma_gemm<false><<<dim3((NH_ * (DN_ + DV_)) / 64, R / 128, 1), blk, 0, stream>>>(
      T3, kv_up_W, T4, R, NH_ * (DN_ + DV_), KVR_, KVR_ + DR_,
      NH_ * (DN_ + DV_), NH_ * (DN_ + DV_),
      0, 0, 0, 0, 0, 0, 1, 1.0f);
  // 7) build per-head Q/K with RoPE
  rope_build_kernel<<<dim3(S_, NH_, B_), 128, 0, stream>>>(T2, T3, T4, pos, Qh, Kh);
  // 8) logits = Q @ K^T / sqrt(192), per (b,h)
  wmma_gemm<true><<<dim3(S_ / 64, S_ / 128, B_ * NH_), blk, 0, stream>>>(
      Qh, Kh, attn, S_, S_, QD_, QD_, QD_, S_,
      (long)NH_ * S_ * QD_, (long)S_ * QD_,
      (long)NH_ * S_ * QD_, (long)S_ * QD_,
      (long)NH_ * S_ * S_,  (long)S_ * S_,
      NH_, 0.07216878364870322f /* 1/sqrt(192) */);
  // 9) softmax in place
  softmax_kernel<<<B_ * NH_ * S_, 256, 0, stream>>>(attn);
  // 10) Cx[b, s, h*128 + d] = attn @ V  (V = T4 cols h*256+128..+256)
  wmma_gemm<false><<<dim3(DV_ / 64, S_ / 128, B_ * NH_), blk, 0, stream>>>(
      attn, T4 + DN_, Cx, S_, DV_, S_, S_, NH_ * (DN_ + DV_), NH_ * DV_,
      (long)NH_ * S_ * S_, (long)S_ * S_,
      (long)S_ * NH_ * (DN_ + DV_), (long)(DN_ + DV_),
      (long)S_ * NH_ * DV_, (long)DV_,
      NH_, 1.0f);
  // 11) out = Cx @ out_W   [4096 x 2048]
  wmma_gemm<false><<<dim3(H_ / 64, R / 128, 1), blk, 0, stream>>>(
      Cx, out_W, out, R, H_, NH_ * DV_, NH_ * DV_, H_, H_,
      0, 0, 0, 0, 0, 0, 1, 1.0f);
}
